// DiTBlock_9259949490455
// MI455X (gfx1250) — compile-verified, matrix operand order checked by reading, not run on hardware
//
#include <hip/hip_runtime.h>
#include <hip/hip_bf16.h>
#include <math.h>

// ---------------------------------------------------------------------------
// DiT block with focused grouped-query linear attention, MI455X (gfx1250).
// All GEMMs via v_wmma_f32_16x16x32_bf16 (fp32 accumulate), wave32.
// B=32 N=1024 HID=1152 HEADS=12 KVH=4 HD=96 KVC=384 MLP=4608
// ---------------------------------------------------------------------------

typedef __bf16 bf16_t;
typedef __attribute__((ext_vector_type(16))) __bf16 v16bf;
typedef __attribute__((ext_vector_type(8)))  __bf16 v8bf;
typedef __attribute__((ext_vector_type(8)))  float  v8f;

__device__ inline v8f wmma_bf16(v16bf a, v16bf b, v8f c) {
#if defined(__HIP_DEVICE_COMPILE__)
  return __builtin_amdgcn_wmma_f32_16x16x32_bf16(false, a, false, b, (short)0, c, false, false);
#else
  return c;
#endif
}

// Load a 16-half A/B fragment from LDS per the CDNA5 16-bit layout:
// lane-half 0 -> K {0..7} and {16..23}; lane-half 1 -> K {8..15} and {24..31}.
// p0 points at K = khalf*8, p1 at K = 16 + khalf*8 (both 16B aligned).
__device__ inline v16bf ld_frag(const bf16_t* p0, const bf16_t* p1) {
  v16bf r;
  ((v8bf*)&r)[0] = *(const v8bf*)p0;
  ((v8bf*)&r)[1] = *(const v8bf*)p1;
  return r;
}

__device__ inline float gelu_tanh(float x) {
  float x3 = x * x * x;
  return 0.5f * x * (1.0f + tanhf(0.7978845608028654f * (x + 0.044715f * x3)));
}

// ---------------------------------------------------------------------------
// Generic bf16 GEMM: C[M x N] = A[M x K] * Bt[N x K]^T (+bias) with epilogues.
// Block tile 128x128, 256 threads (8 waves in 2x4), BK = 32.
// MODE 0: outH = bf16(acc + bias)
// MODE 1: outF = resid + gate[b,col] * (acc + bias)        (proj + residual)
// MODE 2: outH = bf16(gelu(acc + bias))                    (mlp fc1)
// MODE 3: outF += gate[b,col] * acc                        (mlp fc2 accumulate)
// ---------------------------------------------------------------------------
template <int MODE>
__global__ __launch_bounds__(256) void gemm_bf16(
    const bf16_t* __restrict__ A, const bf16_t* __restrict__ Bt,
    const float* __restrict__ bias, int K, int lda, int ldb, int ldc,
    bf16_t* __restrict__ outH, float* __restrict__ outF,
    const float* __restrict__ resid, const float* __restrict__ gate, int gstride) {
  __shared__ bf16_t As[128][40];
  __shared__ bf16_t Bs[128][40];
  const int tid = threadIdx.x;
  const int lane = tid & 31, wave = tid >> 5;
  const int wm = wave >> 2, wn = wave & 3;   // 2 x 4 waves
  const int khalf = lane >> 4, lr = lane & 15;
  const int m0 = blockIdx.x * 128, n0 = blockIdx.y * 128;

  v8f acc[4][2] = {};

  for (int k0 = 0; k0 < K; k0 += 32) {
#pragma unroll
    for (int i = 0; i < 2; ++i) {            // 512 16B chunks per tile, 2/thread
      int c = tid + i * 256;
      int r = c >> 2, cc = c & 3;
      *(uint4*)&As[r][cc * 8] = *(const uint4*)&A[(size_t)(m0 + r) * lda + k0 + cc * 8];
      *(uint4*)&Bs[r][cc * 8] = *(const uint4*)&Bt[(size_t)(n0 + r) * ldb + k0 + cc * 8];
    }
    __syncthreads();
    v16bf bfrag[2];
#pragma unroll
    for (int j = 0; j < 2; ++j) {
      int br = wn * 32 + j * 16 + lr;
      bfrag[j] = ld_frag(&Bs[br][khalf * 8], &Bs[br][16 + khalf * 8]);
    }
#pragma unroll
    for (int i = 0; i < 4; ++i) {
      int ar = wm * 64 + i * 16 + lr;
      v16bf afrag = ld_frag(&As[ar][khalf * 8], &As[ar][16 + khalf * 8]);
#pragma unroll
      for (int j = 0; j < 2; ++j) acc[i][j] = wmma_bf16(afrag, bfrag[j], acc[i][j]);
    }
    __syncthreads();
  }

#pragma unroll
  for (int i = 0; i < 4; ++i) {
#pragma unroll
    for (int j = 0; j < 2; ++j) {
      int rowb = m0 + wm * 64 + i * 16 + 8 * khalf;  // vgpr e holds M = e + 8*half
      int col = n0 + wn * 32 + j * 16 + lr;
#pragma unroll
      for (int e = 0; e < 8; ++e) {
        int r = rowb + e;
        float v = acc[i][j][e];
        if (MODE == 0) {
          v += bias[col];
          outH[(size_t)r * ldc + col] = (bf16_t)v;
        } else if (MODE == 1) {
          v += bias[col];
          int b = r >> 10;
          size_t idx = (size_t)r * ldc + col;
          outF[idx] = resid[idx] + gate[(size_t)b * gstride + col] * v;
        } else if (MODE == 2) {
          v += bias[col];
          outH[(size_t)r * ldc + col] = (bf16_t)gelu_tanh(v);
        } else {  // MODE 3
          int b = r >> 10;
          size_t idx = (size_t)r * ldc + col;
          outF[idx] += gate[(size_t)b * gstride + col] * v;
        }
      }
    }
  }
}

// ---------------------------------------------------------------------------
// Batched kv^T GEMM: per (b,kvh): kvT[96 x 96] = v_t[96 x 1024] * k_t[96 x 1024]^T
// kvT[e][d] = sum_n v[n,e]*khat[n,d]  -> directly the Bt operand of the o-GEMM.
// 192 threads = 6 waves (3m x 2n), each wave 32x48 = 2x3 frags. Output bf16.
// ---------------------------------------------------------------------------
__global__ __launch_bounds__(192) void gemm_kvT(const bf16_t* __restrict__ vt,
                                                const bf16_t* __restrict__ kt,
                                                bf16_t* __restrict__ kvmatT) {
  const int bk = blockIdx.x;  // b*4 + kvh, 128 total
  const bf16_t* A = vt + (size_t)bk * 96 * 1024;
  const bf16_t* B = kt + (size_t)bk * 96 * 1024;
  bf16_t* C = kvmatT + (size_t)bk * 96 * 96;
  __shared__ bf16_t As[96][40];
  __shared__ bf16_t Bs[96][40];
  const int tid = threadIdx.x, lane = tid & 31, wave = tid >> 5;
  const int wm = wave % 3, wn = wave / 3;
  const int khalf = lane >> 4, lr = lane & 15;
  v8f acc[2][3] = {};

  for (int k0 = 0; k0 < 1024; k0 += 32) {
#pragma unroll
    for (int i = 0; i < 2; ++i) {            // 384 chunks, 2/thread
      int c = tid + i * 192;
      int r = c >> 2, cc = c & 3;
      *(uint4*)&As[r][cc * 8] = *(const uint4*)&A[(size_t)r * 1024 + k0 + cc * 8];
      *(uint4*)&Bs[r][cc * 8] = *(const uint4*)&B[(size_t)r * 1024 + k0 + cc * 8];
    }
    __syncthreads();
    v16bf bfrag[3];
#pragma unroll
    for (int j = 0; j < 3; ++j) {
      int br = wn * 48 + j * 16 + lr;
      bfrag[j] = ld_frag(&Bs[br][khalf * 8], &Bs[br][16 + khalf * 8]);
    }
#pragma unroll
    for (int i = 0; i < 2; ++i) {
      int ar = wm * 32 + i * 16 + lr;
      v16bf afrag = ld_frag(&As[ar][khalf * 8], &As[ar][16 + khalf * 8]);
#pragma unroll
      for (int j = 0; j < 3; ++j) acc[i][j] = wmma_bf16(afrag, bfrag[j], acc[i][j]);
    }
    __syncthreads();
  }
#pragma unroll
  for (int i = 0; i < 2; ++i)
#pragma unroll
    for (int j = 0; j < 3; ++j) {
      int rowb = wm * 32 + i * 16 + 8 * khalf;
      int col = wn * 48 + j * 16 + lr;
#pragma unroll
      for (int e = 0; e < 8; ++e) C[(size_t)(rowb + e) * 96 + col] = (bf16_t)acc[i][j][e];
    }
}

// ---------------------------------------------------------------------------
// Batched o GEMM: per (b,h,mtile): o[128 x 96] = qhat[128 x 96] * kvT[96 x 96]^T
// Epilogue adds depthwise-conv term vd (tiled over G: kvh = h % 4) and writes
// the assembled attention tensor attn[b, n, h*96 + d] in bf16 for the proj GEMM.
// 256 threads = 8 waves (4m x 2n), each wave 32x48 = 2x3 frags. K staged once.
// ---------------------------------------------------------------------------
__global__ __launch_bounds__(256) void gemm_o(const bf16_t* __restrict__ qhat,
                                              const bf16_t* __restrict__ kvmatT,
                                              const bf16_t* __restrict__ vd,
                                              bf16_t* __restrict__ attn) {
  const int bh = blockIdx.x;  // b*12 + h, 384 total
  const int mt = blockIdx.y;  // 8 tiles of 128 rows
  const int b = bh / 12, h = bh % 12, kvh = h & 3;
  const bf16_t* A = qhat + (size_t)bh * 1024 * 96 + (size_t)mt * 128 * 96;
  const bf16_t* Bm = kvmatT + (size_t)(b * 4 + kvh) * 96 * 96;
  __shared__ bf16_t As[128][104];
  __shared__ bf16_t Bs[96][104];
  const int tid = threadIdx.x, lane = tid & 31, wave = tid >> 5;
  const int wm = wave >> 1, wn = wave & 1;
  const int khalf = lane >> 4, lr = lane & 15;

  for (int idx = tid; idx < 128 * 12; idx += 256) {
    int r = idx / 12, cc = idx % 12;
    *(uint4*)&As[r][cc * 8] = *(const uint4*)&A[(size_t)r * 96 + cc * 8];
  }
  for (int idx = tid; idx < 96 * 12; idx += 256) {
    int r = idx / 12, cc = idx % 12;
    *(uint4*)&Bs[r][cc * 8] = *(const uint4*)&Bm[(size_t)r * 96 + cc * 8];
  }
  __syncthreads();

  v8f acc[2][3] = {};
#pragma unroll
  for (int s = 0; s < 3; ++s) {
    int k0 = s * 32;
    v16bf bfrag[3];
#pragma unroll
    for (int j = 0; j < 3; ++j) {
      int br = wn * 48 + j * 16 + lr;
      bfrag[j] = ld_frag(&Bs[br][k0 + khalf * 8], &Bs[br][k0 + 16 + khalf * 8]);
    }
#pragma unroll
    for (int i = 0; i < 2; ++i) {
      int ar = wm * 32 + i * 16 + lr;
      v16bf afrag = ld_frag(&As[ar][k0 + khalf * 8], &As[ar][k0 + 16 + khalf * 8]);
#pragma unroll
      for (int j = 0; j < 3; ++j) acc[i][j] = wmma_bf16(afrag, bfrag[j], acc[i][j]);
    }
  }

#pragma unroll
  for (int i = 0; i < 2; ++i)
#pragma unroll
    for (int j = 0; j < 3; ++j) {
      int rowb = mt * 128 + wm * 32 + i * 16 + 8 * khalf;  // n index
      int d = wn * 48 + j * 16 + lr;                       // head-dim index
#pragma unroll
      for (int e = 0; e < 8; ++e) {
        int n = rowb + e;
        float v = acc[i][j][e] +
                  (float)vd[((size_t)(b * 4 + kvh) * 96 + d) * 1024 + n];
        attn[((size_t)(b * 1024 + n)) * 1152 + h * 96 + d] = (bf16_t)v;
      }
    }
}

// ---------------------------------------------------------------------------
// Support kernels
// ---------------------------------------------------------------------------

// fp32 W[K x N] -> bf16 Wt[N x K] (transpose + convert), block (32,8)
__global__ void convert_wT(const float* __restrict__ W, bf16_t* __restrict__ Wt,
                           int K, int N) {
  __shared__ float tile[32][33];
  int n0 = blockIdx.x * 32, k0 = blockIdx.y * 32;
  int tx = threadIdx.x, ty = threadIdx.y;
#pragma unroll
  for (int i = 0; i < 4; ++i)
    tile[ty + i * 8][tx] = W[(size_t)(k0 + ty + i * 8) * N + n0 + tx];
  __syncthreads();
#pragma unroll
  for (int i = 0; i < 4; ++i)
    Wt[(size_t)(n0 + ty + i * 8) * K + k0 + tx] = (bf16_t)tile[tx][ty + i * 8];
}

__global__ void concat_bias(const float* __restrict__ a, const float* __restrict__ b,
                            float* __restrict__ o) {
  int i = blockIdx.x * 256 + threadIdx.x;
  if (i < 1152) o[i] = a[i];
  else if (i < 1920) o[i] = b[i - 1152];
}

// mod[b, 0:6912] = silu(c[b]) @ adaln_w + adaln_b
__global__ __launch_bounds__(128) void adaln_mod(const float* __restrict__ c,
                                                 const float* __restrict__ w,
                                                 const float* __restrict__ bias,
                                                 float* __restrict__ mod) {
  __shared__ float sc[1152];
  int b = blockIdx.y;
  int j = blockIdx.x * 128 + threadIdx.x;
  for (int k = threadIdx.x; k < 1152; k += 128) {
    float x = c[(size_t)b * 1152 + k];
    sc[k] = x / (1.0f + expf(-x));
  }
  __syncthreads();
  float s = bias[j];
  for (int k = 0; k < 1152; ++k) s += sc[k] * w[(size_t)k * 6912 + j];
  mod[(size_t)b * 6912 + j] = s;
}

// out bf16 = layernorm(row) * (1 + mod[b, scaleCol+c]) + mod[b, shiftCol+c]
__global__ __launch_bounds__(256) void ln_modulate(const float* __restrict__ x,
                                                   const float* __restrict__ mod,
                                                   bf16_t* __restrict__ out,
                                                   int shiftCol, int scaleCol) {
  __shared__ float r1[256], r2[256];
  int row = blockIdx.x, b = row >> 10, tid = threadIdx.x;
  const float* xr = x + (size_t)row * 1152;
  float s = 0.f, s2 = 0.f;
  for (int c = tid; c < 1152; c += 256) {
    float v = xr[c];
    s += v;
    s2 += v * v;
  }
  r1[tid] = s; r2[tid] = s2;
  __syncthreads();
  for (int o = 128; o > 0; o >>= 1) {
    if (tid < o) { r1[tid] += r1[tid + o]; r2[tid] += r2[tid + o]; }
    __syncthreads();
  }
  float mean = r1[0] * (1.0f / 1152.0f);
  float var = r2[0] * (1.0f / 1152.0f) - mean * mean;
  float rs = rsqrtf(var + 1e-6f);
  const float* mb = mod + (size_t)b * 6912;
  for (int c = tid; c < 1152; c += 256) {
    float v = (xr[c] - mean) * rs;
    out[(size_t)row * 1152 + c] = (bf16_t)(v * (1.0f + mb[scaleCol + c]) + mb[shiftCol + c]);
  }
}

// Per-(b,head) focused rescale: scale = ||relu(a)||_F / ||relu(a)^3||_F
// groups 0..383 -> q heads, 384..511 -> k kv-heads.
__global__ __launch_bounds__(256) void focused_stats(const bf16_t* __restrict__ qkv,
                                                     float* __restrict__ scales) {
  __shared__ float r1[256], r2[256];
  int g = blockIdx.x, tid = threadIdx.x;
  int b, col0;
  if (g < 384) { b = g / 12; col0 = (g % 12) * 96; }
  else { int k = g - 384; b = k / 4; col0 = 1152 + (k % 4) * 96; }
  const bf16_t* base = qkv + (size_t)b * 1024 * 1920;
  float s2 = 0.f, s6 = 0.f;
  for (int idx = tid; idx < 1024 * 96; idx += 256) {
    int n = idx / 96, d = idx - n * 96;
    float v = fmaxf((float)base[(size_t)n * 1920 + col0 + d], 0.f);
    float v2 = v * v, v3 = v2 * v;
    s2 += v2;
    s6 += v3 * v3;
  }
  r1[tid] = s2; r2[tid] = s6;
  __syncthreads();
  for (int o = 128; o > 0; o >>= 1) {
    if (tid < o) { r1[tid] += r1[tid + o]; r2[tid] += r2[tid + o]; }
    __syncthreads();
  }
  if (tid == 0) {
    float an = sqrtf(r1[0]);
    float apn = sqrtf(r2[0]);
    if (apn == 0.f) apn = 1e-12f;
    scales[g] = an / apn;
  }
}

// Split qkv row -> focused qhat (B,H,N,D), focused k_t (B,KVH,D,N), raw v_t (B,KVH,D,N)
__global__ __launch_bounds__(256) void build_hat(const bf16_t* __restrict__ qkv,
                                                 const float* __restrict__ scales,
                                                 bf16_t* __restrict__ qhat,
                                                 bf16_t* __restrict__ kt,
                                                 bf16_t* __restrict__ vt) {
  int row = blockIdx.x, b = row >> 10, n = row & 1023;
  const bf16_t* src = qkv + (size_t)row * 1920;
  for (int c = threadIdx.x; c < 1920; c += 256) {
    float v = (float)src[c];
    if (c < 1152) {
      int h = c / 96, d = c - h * 96;
      float r = fmaxf(v, 0.f);
      qhat[(((size_t)(b * 12 + h)) * 1024 + n) * 96 + d] =
          (bf16_t)(r * r * r * scales[b * 12 + h]);
    } else if (c < 1536) {
      int cc = c - 1152, kvh = cc / 96, d = cc - kvh * 96;
      float r = fmaxf(v, 0.f);
      kt[(((size_t)(b * 4 + kvh)) * 96 + d) * 1024 + n] =
          (bf16_t)(r * r * r * scales[384 + b * 4 + kvh]);
    } else {
      int cc = c - 1536, kvh = cc / 96, d = cc - kvh * 96;
      vt[(((size_t)(b * 4 + kvh)) * 96 + d) * 1024 + n] = (bf16_t)v;
    }
  }
}

// Depthwise 3x3 SAME conv over v viewed as (B*KVC, 32, 32); one channel / block.
__global__ __launch_bounds__(256) void dwconv3x3(const bf16_t* __restrict__ vt,
                                                 const float* __restrict__ w,
                                                 const float* __restrict__ bias,
                                                 bf16_t* __restrict__ vd) {
  __shared__ float t[34][34];
  int ch = blockIdx.x;           // b*384 + c
  int c = ch % 384;
  for (int i = threadIdx.x; i < 34 * 34; i += 256) {
    int y = i / 34 - 1, x = i % 34 - 1;
    float v = 0.f;
    if ((unsigned)x < 32u && (unsigned)y < 32u)
      v = (float)vt[(size_t)ch * 1024 + y * 32 + x];
    t[i / 34][i % 34] = v;
  }
  __syncthreads();
  float k[9];
#pragma unroll
  for (int i = 0; i < 9; ++i) k[i] = w[c * 9 + i];
  float bb = bias[c];
  for (int p = threadIdx.x; p < 1024; p += 256) {
    int y = p >> 5, x = p & 31;
    float acc = bb;
#pragma unroll
    for (int ky = 0; ky < 3; ++ky)
#pragma unroll
      for (int kx = 0; kx < 3; ++kx) acc += k[ky * 3 + kx] * t[y + ky][x + kx];
    vd[(size_t)ch * 1024 + p] = (bf16_t)acc;
  }
}

// d_out = xnew + gate_mlp * mlp_b2   (MLP chunks then accumulate on top)
__global__ void init_out(const float* __restrict__ xnew, const float* __restrict__ mod,
                         const float* __restrict__ b2, float* __restrict__ out) {
  size_t i = (size_t)blockIdx.x * 256 + threadIdx.x;
  int c = (int)(i % 1152);
  size_t row = i / 1152;
  int b = (int)(row >> 10);
  out[i] = xnew[i] + mod[(size_t)b * 6912 + 5 * 1152 + c] * b2[c];
}

// ---------------------------------------------------------------------------
// Workspace layout (bytes). Peak ~490 MB with region reuse:
//   attn aliases qkv (qkv dead after build_hat), h1 aliases qhat (dead after
//   gemm_o), xnew aliases kt/vt (dead after gemm_kvT/dwconv).
// ---------------------------------------------------------------------------
constexpr size_t AL(size_t x) { return (x + 255) & ~(size_t)255; }
constexpr size_t SZ_QKVT  = 1920ull * 1152 * 2;
constexpr size_t SZ_PROJT = 1152ull * 1152 * 2;
constexpr size_t SZ_W1T   = 4608ull * 1152 * 2;
constexpr size_t SZ_W2T   = 1152ull * 4608 * 2;
constexpr size_t SZ_BIASQ = 1920ull * 4;
constexpr size_t SZ_MOD   = 32ull * 6912 * 4;
constexpr size_t SZ_SCAL  = 512ull * 4;
constexpr size_t SZ_KVMAT = 128ull * 96 * 96 * 2;
constexpr size_t SZ_VD    = 32ull * 384 * 1024 * 2;
constexpr size_t SZ_XM    = 32768ull * 1152 * 2;
constexpr size_t SZ_QKV   = 32768ull * 1920 * 2;
constexpr size_t SZ_QHAT  = 384ull * 1024 * 96 * 2;
constexpr size_t SZ_KT    = 128ull * 96 * 1024 * 2;
constexpr size_t SZ_XNEW  = 32768ull * 1152 * 4;

constexpr size_t O_QKVT  = 0;
constexpr size_t O_PROJT = AL(O_QKVT + SZ_QKVT);
constexpr size_t O_W1T   = AL(O_PROJT + SZ_PROJT);
constexpr size_t O_W2T   = AL(O_W1T + SZ_W1T);
constexpr size_t O_BIASQ = AL(O_W2T + SZ_W2T);
constexpr size_t O_MOD   = AL(O_BIASQ + SZ_BIASQ);
constexpr size_t O_SCAL  = AL(O_MOD + SZ_MOD);
constexpr size_t O_KVMAT = AL(O_SCAL + SZ_SCAL);
constexpr size_t O_VD    = AL(O_KVMAT + SZ_KVMAT);
constexpr size_t O_XM    = AL(O_VD + SZ_VD);
constexpr size_t O_QKV   = AL(O_XM + SZ_XM);
constexpr size_t O_QHAT  = AL(O_QKV + SZ_QKV);
constexpr size_t O_KT    = AL(O_QHAT + SZ_QHAT);
constexpr size_t O_VT    = AL(O_KT + SZ_KT);
constexpr size_t O_ATTN  = O_QKV;   // reuse
constexpr size_t O_H1    = O_QHAT;  // reuse
constexpr size_t O_XNEW  = O_KT;    // reuse (needs SZ_XNEW past O_KT)

extern "C" void kernel_launch(void* const* d_in, const int* in_sizes, int n_in,
                              void* d_out, int out_size, void* d_ws, size_t ws_size,
                              hipStream_t stream) {
  (void)in_sizes; (void)n_in; (void)out_size; (void)ws_size;
  const float* x      = (const float*)d_in[0];
  const float* cvec   = (const float*)d_in[1];
  const float* wq_w   = (const float*)d_in[2];
  const float* wq_b   = (const float*)d_in[3];
  const float* wkv_w  = (const float*)d_in[4];
  const float* wkv_b  = (const float*)d_in[5];
  const float* dwc_w  = (const float*)d_in[6];
  const float* dwc_b  = (const float*)d_in[7];
  const float* proj_w = (const float*)d_in[8];
  const float* proj_b = (const float*)d_in[9];
  const float* adaln_w = (const float*)d_in[10];
  const float* adaln_b = (const float*)d_in[11];
  const float* mlp_w1 = (const float*)d_in[12];
  const float* mlp_b1 = (const float*)d_in[13];
  const float* mlp_w2 = (const float*)d_in[14];
  const float* mlp_b2 = (const float*)d_in[15];
  float* out = (float*)d_out;

  char* ws = (char*)d_ws;
  bf16_t* qkvT  = (bf16_t*)(ws + O_QKVT);
  bf16_t* projT = (bf16_t*)(ws + O_PROJT);
  bf16_t* w1T   = (bf16_t*)(ws + O_W1T);
  bf16_t* w2T   = (bf16_t*)(ws + O_W2T);
  float*  biasq = (float*)(ws + O_BIASQ);
  float*  mod   = (float*)(ws + O_MOD);
  float*  scal  = (float*)(ws + O_SCAL);
  bf16_t* kvmat = (bf16_t*)(ws + O_KVMAT);
  bf16_t* vd    = (bf16_t*)(ws + O_VD);
  bf16_t* xm    = (bf16_t*)(ws + O_XM);
  bf16_t* qkv   = (bf16_t*)(ws + O_QKV);
  bf16_t* qhat  = (bf16_t*)(ws + O_QHAT);
  bf16_t* kt    = (bf16_t*)(ws + O_KT);
  bf16_t* vt    = (bf16_t*)(ws + O_VT);
  bf16_t* attn  = (bf16_t*)(ws + O_ATTN);
  bf16_t* h1    = (bf16_t*)(ws + O_H1);
  float*  xnew  = (float*)(ws + O_XNEW);

  const dim3 tb(32, 8);
  // 1) weight convert/transpose to bf16
  convert_wT<<<dim3(1152 / 32, 1152 / 32), tb, 0, stream>>>(wq_w, qkvT, 1152, 1152);
  convert_wT<<<dim3(768 / 32, 1152 / 32), tb, 0, stream>>>(wkv_w, qkvT + 1152ull * 1152, 1152, 768);
  convert_wT<<<dim3(1152 / 32, 1152 / 32), tb, 0, stream>>>(proj_w, projT, 1152, 1152);
  convert_wT<<<dim3(4608 / 32, 1152 / 32), tb, 0, stream>>>(mlp_w1, w1T, 1152, 4608);
  convert_wT<<<dim3(1152 / 32, 4608 / 32), tb, 0, stream>>>(mlp_w2, w2T, 4608, 1152);
  concat_bias<<<8, 256, 0, stream>>>(wq_b, wkv_b, biasq);

  // 2) adaLN modulation vector
  adaln_mod<<<dim3(54, 32), 128, 0, stream>>>(cvec, adaln_w, adaln_b, mod);

  // 3) xm = modulate(layernorm(x), shift_msa, scale_msa)  [bf16]
  ln_modulate<<<32768, 256, 0, stream>>>(x, mod, xm, 0, 1152);

  // 4) qkv = xm @ [wq|wkv] + bias  [bf16, 32768 x 1920]
  gemm_bf16<0><<<dim3(256, 15), 256, 0, stream>>>(xm, qkvT, biasq, 1152, 1152, 1152,
                                                  1920, qkv, nullptr, nullptr, nullptr, 0);

  // 5) focused rescale factors, then split into qhat / k_t / v_t
  focused_stats<<<512, 256, 0, stream>>>(qkv, scal);
  build_hat<<<32768, 256, 0, stream>>>(qkv, scal, qhat, kt, vt);

  // 6) depthwise 3x3 conv on v
  dwconv3x3<<<32 * 384, 256, 0, stream>>>(vt, dwc_w, dwc_b, vd);

  // 7) kvT[b,kvh] = v^T * khat  (bf16, already transposed for next GEMM)
  gemm_kvT<<<128, 192, 0, stream>>>(vt, kt, kvmat);

  // 8) o = qhat @ kv (+ dwconv term), assembled as attn[b,n,1152] bf16
  gemm_o<<<dim3(384, 8), 256, 0, stream>>>(qhat, kvmat, vd, attn);

  // 9) xnew = x + gate_msa * (attn @ proj_w + proj_b)
  gemm_bf16<1><<<dim3(256, 9), 256, 0, stream>>>(attn, projT, proj_b, 1152, 1152, 1152,
                                                 1152, nullptr, xnew, x, mod + 2 * 1152, 6912);

  // 10) xm = modulate(layernorm(xnew), shift_mlp, scale_mlp)
  ln_modulate<<<32768, 256, 0, stream>>>(xnew, mod, xm, 3 * 1152, 4 * 1152);

  // 11) d_out = xnew + gate_mlp * b2; then accumulate MLP in 4 chunks of 1152
  init_out<<<147456, 256, 0, stream>>>(xnew, mod, mlp_b2, out);
  for (int j = 0; j < 4; ++j) {
    gemm_bf16<2><<<dim3(256, 9), 256, 0, stream>>>(
        xm, w1T + (size_t)j * 1152 * 1152, mlp_b1 + j * 1152, 1152, 1152, 1152, 1152,
        h1, nullptr, nullptr, nullptr, 0);
    gemm_bf16<3><<<dim3(256, 9), 256, 0, stream>>>(
        h1, w2T + (size_t)j * 1152, nullptr, 1152, 1152, 4608, 1152,
        nullptr, out, nullptr, mod + 5 * 1152, 6912);
  }
}
